// SlotAttention_60610578481205
// MI455X (gfx1250) — matrix operand order repaired and checked by reading, not compiled, not run on hardware
//
#include <hip/hip_runtime.h>
#include <hip/hip_bf16.h>
#include <math.h>

typedef __bf16 bf16;
typedef __attribute__((ext_vector_type(16))) __bf16 v16bf;
typedef __attribute__((ext_vector_type(8)))  float  v8f;

#define B_ 64
#define N_ 4096
#define D_ 64
#define S_ 7
#define NUM_ITER 3
#define EPS_ATTN 1e-8f
#define LN_EPS   1e-5f
#define SCALE    0.125f   // 1/sqrt(64)

// workspace layout (bytes)
#define KB_OFF   0ull                      // k bf16 [B*N][64]  = 32 MB
#define VB_OFF   33554432ull               // v bf16 [B*N][64]  = 32 MB
#define QW_OFF   67108864ull               // q bf16 [B][16][64] padded slots
#define NUM_OFF  (QW_OFF + 131072ull)      // numer f32 [B][7][64]
#define DEN_OFF  (NUM_OFF + 114688ull)     // denom f32 [B][7]
#define WKB_OFF  (DEN_OFF + 4096ull)       // Wk^T bf16 [64][64]
#define WVB_OFF  (WKB_OFF + 8192ull)       // Wv^T bf16 [64][64]

// CDNA5 16-bit A-matrix (16x32) K packing: VGPR v, lane-half h -> K pair base
__device__ __forceinline__ int kbA(int v, int h) { return (v < 4 ? 2*v : 16 + 2*(v-4)) + 8*h; }
// CDNA5 16-bit B-matrix (32x16) K packing: same K order as A (VGPR v, lane-half h -> K pair base)
__device__ __forceinline__ int kbB(int v, int h) { return (v < 4 ? 2*v : 16 + 2*(v-4)) + 8*h; }

union U16 { v16bf v; unsigned int u[8]; };

__device__ __forceinline__ v8f wmma_bf16(v16bf a, v16bf b, v8f c) {
  return __builtin_amdgcn_wmma_f32_16x16x32_bf16(false, a, false, b, (short)0, c, false, false);
}

// DS ops from one wave execute in order; this drains DScnt and pins compiler
// ordering for the cross-lane LDS handoff within a wave (no block barrier needed:
// each wave owns a private LDS region).
__device__ __forceinline__ void wave_lds_fence() {
  asm volatile("s_wait_dscnt 0" ::: "memory");
}

// ---------------------------------------------------------------------------
// Kernel 0: convert Wk/Wv to transposed bf16 (so B-matrix K-pairs are dword
// contiguous); broadcast slots_init into the slots buffer (d_out).
__global__ void k_prep(const float* __restrict__ Wk, const float* __restrict__ Wv,
                       const float* __restrict__ sl0,
                       bf16* __restrict__ wkb, bf16* __restrict__ wvb,
                       float* __restrict__ slots) {
  int idx = blockIdx.x * blockDim.x + threadIdx.x;
  if (idx < 4096) {
    int din = idx >> 6, dout = idx & 63;
    wkb[dout*64 + din] = (bf16)Wk[din*64 + dout];
    wvb[dout*64 + din] = (bf16)Wv[din*64 + dout];
  }
  for (int i = idx; i < B_*S_*D_; i += gridDim.x*blockDim.x)
    slots[i] = sl0[i % (S_*D_)];
}

// ---------------------------------------------------------------------------
// Kernel 1: fused LayerNorm(x) + k = xn@Wk, v = xn@Wv  (bf16 WMMA, f32 acc)
__global__ __launch_bounds__(128) void k_lnproj(
    const float* __restrict__ x, const float* __restrict__ lg,
    const float* __restrict__ lb, const bf16* __restrict__ wkb,
    const bf16* __restrict__ wvb, bf16* __restrict__ kb, bf16* __restrict__ vb) {
  __shared__ float xs[4][16][64];
  __shared__ bf16  xn[4][16][64];
  const int wave = threadIdx.x >> 5, lane = threadIdx.x & 31;
  const int m = lane & 15, h = lane >> 4;

  // Hoist weight B-matrices (tile-invariant): 4 output groups x 2 K-halves each
  U16 bk[4][2], bw[4][2];
  {
    const unsigned int* wkp = (const unsigned int*)wkb;
    const unsigned int* wvp = (const unsigned int*)wvb;
    for (int g = 0; g < 4; ++g) {
      int dout = g*16 + m;
      for (int kk = 0; kk < 2; ++kk)
        for (int v = 0; v < 8; ++v) {
          int kd = kbB(v, h) + 32*kk;
          bk[g][kk].u[v] = wkp[dout*32 + (kd >> 1)];
          bw[g][kk].u[v] = wvp[dout*32 + (kd >> 1)];
        }
    }
  }

  const int tstep = gridDim.x*4;
  for (int t = blockIdx.x*4 + wave; t < (B_*N_)/16; t += tstep) {
    const long tok0 = (long)t * 16;
    const float* xp = x + tok0*64;
    // prefetch next tile (4 KB = 32 x 128B lines; one line per lane)
    if (t + tstep < (B_*N_)/16)
      __builtin_prefetch(x + ((long)(t + tstep)*16)*64 + lane*32, 0, 1);
    for (int j = 0; j < 32; ++j) {              // stage 16x64 f32 tile
      int idx = j*32 + lane;
      xs[wave][idx >> 6][idx & 63] = xp[idx];
    }
    asm volatile("" ::: "memory");
    if (lane < 16) {                            // per-row LayerNorm -> bf16
      float mn = 0.f;
      for (int c = 0; c < 64; ++c) mn += xs[wave][lane][c];
      mn *= (1.f/64.f);
      float var = 0.f;
      for (int c = 0; c < 64; ++c) { float d = xs[wave][lane][c]-mn; var += d*d; }
      float rs = rsqrtf(var*(1.f/64.f) + LN_EPS);
      for (int c = 0; c < 64; ++c)
        xn[wave][lane][c] = (bf16)((xs[wave][lane][c]-mn)*rs*lg[c] + lb[c]);
    }
    wave_lds_fence();
    U16 alo, ahi;                               // A = xn tile (M=token,K=din)
    const unsigned int* xr = (const unsigned int*)&xn[wave][m][0];
    for (int v = 0; v < 8; ++v) {
      int kd = kbA(v, h);
      alo.u[v] = xr[kd >> 1];
      ahi.u[v] = xr[(kd + 32) >> 1];
    }
    for (int g = 0; g < 4; ++g) {
      v8f ck = {}; v8f cv = {};
      ck = wmma_bf16(alo.v, bk[g][0].v, ck);
      ck = wmma_bf16(ahi.v, bk[g][1].v, ck);
      cv = wmma_bf16(alo.v, bw[g][0].v, cv);
      cv = wmma_bf16(ahi.v, bw[g][1].v, cv);
      int dim = g*16 + m;
      for (int j = 0; j < 8; ++j) {             // C: row M=j(+8 for upper lanes)
        long row = tok0 + j + 8*h;
        kb[row*64 + dim] = (bf16)ck[j];
        vb[row*64 + dim] = (bf16)cv[j];
      }
    }
    asm volatile("" ::: "memory");
  }
}

// ---------------------------------------------------------------------------
// Kernel 2 (per iter): zero accumulators; sn = LN(slots); q = sn@Wq -> bf16,
// stored padded to 16 slot rows (rows 7..15 zero) so q is a ready WMMA A-matrix.
__global__ __launch_bounds__(64) void k_qprep(
    const float* __restrict__ slots, const float* __restrict__ lsg,
    const float* __restrict__ lsb, const float* __restrict__ Wq,
    bf16* __restrict__ qw, float* __restrict__ numer, float* __restrict__ denom) {
  __shared__ float sn[S_][64];
  __shared__ float mrs[S_][2];
  int t = threadIdx.x, b = blockIdx.x;
  for (int i = t; i < S_*D_; i += 64) numer[b*S_*D_ + i] = 0.f;
  if (t < S_) denom[b*S_ + t] = 0.f;
  const float* sp = slots + b*S_*D_;
  if (t < S_) {
    float mn = 0.f;
    for (int c = 0; c < 64; ++c) mn += sp[t*64+c];
    mn *= (1.f/64.f);
    float var = 0.f;
    for (int c = 0; c < 64; ++c) { float d = sp[t*64+c]-mn; var += d*d; }
    mrs[t][0] = mn; mrs[t][1] = rsqrtf(var*(1.f/64.f) + LN_EPS);
  }
  __syncthreads();
  for (int s = 0; s < S_; ++s)
    sn[s][t] = (sp[s*64+t]-mrs[s][0])*mrs[s][1]*lsg[t] + lsb[t];
  __syncthreads();
  float q[S_] = {};
  for (int k = 0; k < 64; ++k) {
    float w = Wq[k*64 + t];
    for (int s = 0; s < S_; ++s) q[s] += sn[s][k]*w;
  }
  bf16* qb = qw + b*1024;
  for (int s = 0; s < S_; ++s) qb[s*64 + t] = (bf16)q[s];
  for (int s = S_; s < 16; ++s) qb[s*64 + t] = (bf16)0.f;
}

// ---------------------------------------------------------------------------
// Kernel 3 (per iter): per 32-token chunk:
//   logits (WMMA, M=slots N=tokens -> softmax over slots is lane-local),
//   p -> LDS, numer += P^T @ v (WMMA K=32 tokens), denom via shfl + atomics.
__global__ __launch_bounds__(128) void k_attn(
    const bf16* __restrict__ kbuf, const bf16* __restrict__ vbuf,
    const bf16* __restrict__ qw, float* __restrict__ numer,
    float* __restrict__ denom) {
  __shared__ bf16 k_s[4][32][64];   // natural: token-major (K-dim pairs contiguous)
  __shared__ bf16 v_t[4][64][32];   // transposed: dim-major (token pairs contiguous)
  __shared__ bf16 p_b[4][16][32];   // p: slot-major (token pairs contiguous)
  const int wave = threadIdx.x >> 5, lane = threadIdx.x & 31;
  const int m = lane & 15, h = lane >> 4;
  const int b = blockIdx.x >> 2, seg = blockIdx.x & 3;

  U16 alo, ahi;                     // A = q (M=slot, K=dim), K=64 -> two halves
  {
    const unsigned int* qp = (const unsigned int*)(qw + b*1024 + m*64);
    for (int v = 0; v < 8; ++v) {
      int kd = kbA(v, h);
      alo.u[v] = qp[kd >> 1];
      ahi.u[v] = qp[(kd + 32) >> 1];
    }
  }
  v8f acc[4] = {};                  // numer accumulators, 4 dim-groups of 16
  float dden[S_] = {};

  // prefetch first chunk (4 KB each of k,v; one 128B line per lane)
  {
    const long g0 = ((long)b*N_ + seg*1024 + wave*8*32) * 64;
    __builtin_prefetch(kbuf + g0 + lane*64, 0, 1);
    __builtin_prefetch(vbuf + g0 + lane*64, 0, 1);
  }

  for (int i = 0; i < 8; ++i) {
    const long gtok = (long)b*N_ + seg*1024 + (wave*8 + i)*32;
    const unsigned int* kg = (const unsigned int*)(kbuf + gtok*64);
    const unsigned int* vg = (const unsigned int*)(vbuf + gtok*64);
    if (i < 7) {                    // prefetch next chunk while this one computes
      __builtin_prefetch(kbuf + (gtok + 32)*64 + lane*64, 0, 1);
      __builtin_prefetch(vbuf + (gtok + 32)*64 + lane*64, 0, 1);
    }
    unsigned int* ks = (unsigned int*)&k_s[wave][0][0];
    for (int j = 0; j < 32; ++j) {
      int idx = j*32 + lane;
      ks[idx] = kg[idx];
      union { unsigned int u; bf16 hh[2]; } cv; cv.u = vg[idx];
      int tok = idx >> 5, d0 = (idx & 31) * 2;
      v_t[wave][d0][tok]   = cv.hh[0];
      v_t[wave][d0+1][tok] = cv.hh[1];
    }
    wave_lds_fence();
    for (int g = 0; g < 2; ++g) {   // two 16-token groups: logits + softmax
      int tok = g*16 + m;
      const unsigned int* kr = (const unsigned int*)&k_s[wave][tok][0];
      U16 blo, bhi;
      for (int v = 0; v < 8; ++v) {
        int kd = kbB(v, h);
        blo.u[v] = kr[kd >> 1];
        bhi.u[v] = kr[(kd + 32) >> 1];
      }
      v8f c = {};
      c = wmma_bf16(alo.v, blo.v, c);
      c = wmma_bf16(ahi.v, bhi.v, c);
      if (lane < 16) {              // lane = token; VGPRs 0..6 = slot logits
        float l[S_], mx = -1e30f;
        for (int s = 0; s < S_; ++s) { l[s] = c[s]*SCALE; mx = fmaxf(mx, l[s]); }
        float sum = 0.f;
        for (int s = 0; s < S_; ++s) { l[s] = __expf(l[s]-mx); sum += l[s]; }
        float inv = 1.f/sum;
        for (int s = 0; s < S_; ++s) {
          float p = l[s]*inv + EPS_ATTN;
          dden[s] += p;
          p_b[wave][s][g*16 + lane] = (bf16)p;
        }
      }
    }
    wave_lds_fence();
    U16 ap;                         // A = P^T (M=slot, K=32 tokens)
    {
      const unsigned int* pr = (const unsigned int*)&p_b[wave][m][0];
      for (int v = 0; v < 8; ++v) ap.u[v] = pr[kbA(v, h) >> 1];
    }
    for (int gg = 0; gg < 4; ++gg) {
      int dim = gg*16 + m;
      const unsigned int* vr = (const unsigned int*)&v_t[wave][dim][0];
      U16 bv;
      for (int v = 0; v < 8; ++v) bv.u[v] = vr[kbB(v, h) >> 1];
      acc[gg] = wmma_bf16(ap.v, bv.v, acc[gg]);
    }
    wave_lds_fence();
  }
  if (lane < 16) {                  // C rows 0..6 = slots, N=lane = dim
    for (int gg = 0; gg < 4; ++gg)
      for (int j = 0; j < S_; ++j)
        atomicAdd(&numer[b*S_*D_ + j*64 + gg*16 + lane], acc[gg][j]);
  }
  for (int s = 0; s < S_; ++s) {
    float v = dden[s];
    for (int off = 16; off > 0; off >>= 1) v += __shfl_down(v, off);
    if (lane == 0) atomicAdd(&denom[b*S_ + s], v);
  }
}

// ---------------------------------------------------------------------------
// Kernel 4 (per iter): updates = numer/denom; GRU cell; MLP residual. Tiny.
__global__ __launch_bounds__(64) void k_update(
    float* __restrict__ slots, const float* __restrict__ numer,
    const float* __restrict__ denom, const float* __restrict__ wih,
    const float* __restrict__ whh, const float* __restrict__ bih,
    const float* __restrict__ bhh, const float* __restrict__ w1,
    const float* __restrict__ b1, const float* __restrict__ w2,
    const float* __restrict__ b2) {
  __shared__ float u[S_][64], hh[S_][64], hn[S_][64], mm[S_][128];
  int t = threadIdx.x, b = blockIdx.x;
  for (int s = 0; s < S_; ++s) {
    u[s][t]  = numer[b*S_*D_ + s*64 + t] / denom[b*S_ + s];
    hh[s][t] = slots[b*S_*D_ + s*64 + t];
  }
  __syncthreads();
  float bir = bih[t], biz = bih[64+t], bin = bih[128+t];
  float bhr = bhh[t], bhz = bhh[64+t], bhn = bhh[128+t];
  for (int s = 0; s < S_; ++s) {
    float gir=bir, giz=biz, gin=bin, ghr=bhr, ghz=bhz, ghn=bhn;
    for (int k = 0; k < 64; ++k) {
      float uk = u[s][k], hk = hh[s][k];
      gir += uk*wih[t*64+k];        giz += uk*wih[(64+t)*64+k];
      gin += uk*wih[(128+t)*64+k];  ghr += hk*whh[t*64+k];
      ghz += hk*whh[(64+t)*64+k];   ghn += hk*whh[(128+t)*64+k];
    }
    float r = 1.f/(1.f+__expf(-(gir+ghr)));
    float z = 1.f/(1.f+__expf(-(giz+ghz)));
    float n = tanhf(gin + r*ghn);
    hn[s][t] = (1.f-z)*n + z*hh[s][t];
  }
  __syncthreads();
  for (int s = 0; s < S_; ++s)
    for (int jj = t; jj < 128; jj += 64) {
      float a = b1[jj];
      for (int k = 0; k < 64; ++k) a += hn[s][k]*w1[k*128+jj];
      mm[s][jj] = fmaxf(a, 0.f);
    }
  __syncthreads();
  for (int s = 0; s < S_; ++s) {
    float a = b2[t];
    for (int j = 0; j < 128; ++j) a += mm[s][j]*w2[j*64+t];
    slots[b*S_*D_ + s*64 + t] = hn[s][t] + a;
  }
}

// ---------------------------------------------------------------------------
extern "C" void kernel_launch(void* const* d_in, const int* in_sizes, int n_in,
                              void* d_out, int out_size, void* d_ws, size_t ws_size,
                              hipStream_t stream) {
  const float* x   = (const float*)d_in[0];
  const float* lig = (const float*)d_in[1];
  const float* lib = (const float*)d_in[2];
  const float* lsg = (const float*)d_in[3];
  const float* lsb = (const float*)d_in[4];
  const float* sl0 = (const float*)d_in[5];
  const float* Wk  = (const float*)d_in[6];
  const float* Wv  = (const float*)d_in[7];
  const float* Wq  = (const float*)d_in[8];
  const float* wih = (const float*)d_in[9];
  const float* whh = (const float*)d_in[10];
  const float* bih = (const float*)d_in[11];
  const float* bhh = (const float*)d_in[12];
  const float* w1  = (const float*)d_in[13];
  const float* b1  = (const float*)d_in[14];
  const float* w2  = (const float*)d_in[15];
  const float* b2  = (const float*)d_in[16];

  char* ws = (char*)d_ws;
  bf16*  kb  = (bf16*)(ws + KB_OFF);
  bf16*  vb  = (bf16*)(ws + VB_OFF);
  bf16*  qw  = (bf16*)(ws + QW_OFF);
  float* nm  = (float*)(ws + NUM_OFF);
  float* dn  = (float*)(ws + DEN_OFF);
  bf16*  wkb = (bf16*)(ws + WKB_OFF);
  bf16*  wvb = (bf16*)(ws + WVB_OFF);
  float* slots = (float*)d_out;

  k_prep  <<<16, 256, 0, stream>>>(Wk, Wv, sl0, wkb, wvb, slots);
  k_lnproj<<<512, 128, 0, stream>>>(x, lig, lib, wkb, wvb, kb, vb);
  for (int it = 0; it < NUM_ITER; ++it) {
    k_qprep <<<B_, 64, 0, stream>>>(slots, lsg, lsb, Wq, qw, nm, dn);
    k_attn  <<<B_*4, 128, 0, stream>>>(kb, vb, qw, nm, dn);
    k_update<<<B_, 64, 0, stream>>>(slots, nm, dn, wih, whh, bih, bhh,
                                    w1, b1, w2, b2);
  }
}
